// ModelWrapper_13176959664675
// MI455X (gfx1250) — matrix operand order repaired and checked by reading, not compiled, not run on hardware
//
#include <hip/hip_runtime.h>
#include <hip/hip_bf16.h>
#include <stdint.h>

// Problem constants
#define S_ 384
#define H_ 12
#define V_ 64
#define D_ 768

typedef __attribute__((ext_vector_type(16))) __bf16 v16bf;
typedef __attribute__((ext_vector_type(8)))  float  v8f;

union ABu { v16bf v; unsigned short u[16]; };
union BFu { __bf16 b; unsigned short u; };

__device__ __forceinline__ __bf16 tobf(float f) { return (__bf16)f; }  // fptrunc, RNE

__device__ __forceinline__ unsigned short bf_bits(float f) {
  BFu c; c.b = (__bf16)f; return c.u;
}

// -------- small prep kernels --------------------------------------------

// wbar[h*V+v] = mean over d of dense_w[d,h,v]
__global__ void k_wbar(const float* __restrict__ w, float* __restrict__ wbar) {
  int idx = blockIdx.x * blockDim.x + threadIdx.x;
  if (idx >= H_ * V_) return;
  float s = 0.f;
  for (int dd = 0; dd < D_; ++dd) s += w[(size_t)dd * (H_ * V_) + idx];
  wbar[idx] = s * (1.0f / D_);
}

// meanT[h*S+s] = mean over d of transformed[h,s,:]
__global__ void k_meanT(const float* __restrict__ val, const float* __restrict__ wbar,
                        float* __restrict__ meanT) {
  int idx = blockIdx.x * blockDim.x + threadIdx.x;
  if (idx >= H_ * S_) return;
  int h = idx / S_, s = idx % S_;
  const float* vp = val + (size_t)(h * S_ + s) * V_;
  const float* wp = wbar + h * V_;
  float acc = 0.f;
  for (int v = 0; v < V_; ++v) acc += vp[v] * wp[v];
  meanT[idx] = acc;
}

// out[row] = mean over D of x[row,:]   (wave per row)
__global__ void k_rowmean(const float* __restrict__ x, float* __restrict__ out) {
  int lane = threadIdx.x & 31;
  int row  = blockIdx.x * (blockDim.x >> 5) + (threadIdx.x >> 5);
  if (row >= S_) return;
  const float* p = x + (size_t)row * D_;
  float s = 0.f;
  for (int i = lane; i < D_; i += 32) s += p[i];
  #pragma unroll
  for (int m = 16; m; m >>= 1) s += __shfl_xor(s, m, 32);
  if (lane == 0) out[row] = s * (1.0f / D_);
}

// out[row] = 1 / std(x[row,:], ddof=1)   (wave per row)
__global__ void k_invstd(const float* __restrict__ x, float* __restrict__ out) {
  int lane = threadIdx.x & 31;
  int row  = blockIdx.x * (blockDim.x >> 5) + (threadIdx.x >> 5);
  if (row >= S_) return;
  const float* p = x + (size_t)row * D_;
  float s = 0.f, q = 0.f;
  for (int i = lane; i < D_; i += 32) { float v = p[i]; s += v; q += v * v; }
  #pragma unroll
  for (int m = 16; m; m >>= 1) { s += __shfl_xor(s, m, 32); q += __shfl_xor(q, m, 32); }
  if (lane == 0) {
    float mean = s * (1.0f / D_);
    float var  = (q - (float)D_ * mean * mean) * (1.0f / (D_ - 1));
    out[row] = 1.0f / sqrtf(var);
  }
}

// bias_term[d] = (bias[d] - mean(bias)) * ln_weight[d]
__global__ void k_bias(const float* __restrict__ bias, const float* __restrict__ lnw,
                       float* __restrict__ bias_term) {
  __shared__ float red[256];
  int t = threadIdx.x;
  red[t] = bias[t] + bias[t + 256] + bias[t + 512];
  __syncthreads();
  for (int off = 128; off; off >>= 1) {
    if (t < off) red[t] += red[t + off];
    __syncthreads();
  }
  float bmean = red[0] * (1.0f / D_);
  for (int dd = t; dd < D_; dd += 256)
    bias_term[dd] = (bias[dd] - bmean) * lnw[dd];
}

// -------- GEMM 1: transformed[h,s,d] = sum_v value[h,s,v]*W[d,h,v] -------
// bf16 WMMA 16x16x32, K=v in two steps. Output stored as [s][d][h padded to 16]
// so the second GEMM's B operand (K = h) is 6 contiguous dwords per (s,d).
__global__ void k_transformed(const float* __restrict__ val,
                              const float* __restrict__ w,
                              unsigned short* __restrict__ Tst) {
  const int lane = threadIdx.x & 31;
  const int wave = threadIdx.x >> 5;
  const int h    = blockIdx.x;
  const int d0   = (blockIdx.y * 8 + wave) * 16;
  const int n    = lane & 15, half = lane >> 4;

  // B (K=v, N=d): VGPR j -> K = (j/4)*16 + half*8 + (j%4)*2
  ABu b0, b1;
  const float* wp = w + (size_t)(d0 + n) * (H_ * V_) + h * V_;
  #pragma unroll
  for (int j = 0; j < 8; ++j) {
    int K = ((j >> 2) * 16) + half * 8 + (j & 3) * 2;
    b0.v[2*j]   = tobf(wp[K]);
    b0.v[2*j+1] = tobf(wp[K + 1]);
    b1.v[2*j]   = tobf(wp[32 + K]);
    b1.v[2*j+1] = tobf(wp[32 + K + 1]);
  }

  for (int st = 0; st < S_ / 16; ++st) {
    const int s0 = st * 16;
    const float* vp = val + (size_t)(h * S_ + s0 + n) * V_;
    // A (M=s, K=v): lane M = lane%16; VGPR j -> K = (j/4)*16 + half*8 + (j%4)*2
    ABu a0, a1;
    #pragma unroll
    for (int j = 0; j < 8; ++j) {
      int K = ((j >> 2) * 16) + half * 8 + (j & 3) * 2;
      a0.v[2*j]   = tobf(vp[K]);
      a0.v[2*j+1] = tobf(vp[K + 1]);
      a1.v[2*j]   = tobf(vp[32 + K]);
      a1.v[2*j+1] = tobf(vp[32 + K + 1]);
    }
    v8f c = {};
    c = __builtin_amdgcn_wmma_f32_16x16x32_bf16(false, a0.v, false, b0.v, (short)0, c, false, false);
    c = __builtin_amdgcn_wmma_f32_16x16x32_bf16(false, a1.v, false, b1.v, (short)0, c, false, false);
    // C layout: VGPR r, lanes0-15 -> M=r, lanes16-31 -> M=r+8 ; N = lane%16
    #pragma unroll
    for (int r = 0; r < 8; ++r) {
      int srow = s0 + r + 8 * half;
      Tst[((size_t)srow * D_ + d0 + n) * 16 + h] = bf_bits(c[r]);
    }
  }
}

// -------- Main fused pass: summed + residual + center*gamma + scales -----
// Wave owns a (k-tile, d-tile), loops all s: one WMMA (K=h padded) per s,
// writes tv_std, accumulates attn_output in registers. The shared attention
// tile is staged with CDNA5 async global->LDS DMA (ASYNCcnt tracked).
__global__ void k_main(const float* __restrict__ attn,
                       const float* __restrict__ hidden,
                       const unsigned short* __restrict__ Tst,
                       const float* __restrict__ meanT,
                       const float* __restrict__ meanH,
                       const float* __restrict__ inv_std,
                       const float* __restrict__ ln_w,
                       float* __restrict__ out_tvstd,
                       float* __restrict__ ws_attnout) {
  __shared__ float lds_attn[H_ * 16];    // attn[h][k0+m] for current s
  const int tid  = threadIdx.x;
  const int lane = tid & 31, wave = tid >> 5;
  const int k0   = blockIdx.x * 16;
  const int d0   = (blockIdx.y * 8 + wave) * 16;
  const int n    = lane & 15, half = lane >> 4;
  const int d    = d0 + n;

  const float lnw = ln_w[d];
  float invs[8], acc[8];
  #pragma unroll
  for (int r = 0; r < 8; ++r) { invs[r] = inv_std[k0 + r + 8 * half]; acc[r] = 0.f; }

  // per-thread source row for the staged attention element (h = tid/16, k = k0+tid%16)
  const float* attn_src = attn + ((size_t)(tid >> 4) * S_ + k0 + (tid & 15)) * S_;
  const unsigned lds_dst = (unsigned)(uintptr_t)&lds_attn[tid];   // LDS addr = low 32 bits

  for (int s = 0; s < S_; ++s) {
    __syncthreads();                      // previous iteration done reading LDS
    if (tid < H_ * 16) {
      // GLOBAL_LOAD_ASYNC_TO_LDS_B32: memory -> LDS directly, ASYNCcnt tracked
      asm volatile("global_load_async_to_lds_b32 %0, %1, off"
                   :: "v"(lds_dst), "v"(attn_src + s) : "memory");
    }
#if __has_builtin(__builtin_amdgcn_s_wait_asynccnt)
    __builtin_amdgcn_s_wait_asynccnt(0);
#else
    asm volatile("s_wait_asynccnt 0" ::: "memory");
#endif
    __syncthreads();                      // publish staged tile to all 8 waves

    // A (M=k, K=h padded to 32 with zeros)
    ABu a, b;
    #pragma unroll
    for (int j = 0; j < 8; ++j) {
      int K = half * 16 + 2 * j;
      a.v[2*j]   = (K     < H_) ? tobf(lds_attn[K * 16 + n])       : tobf(0.f);
      a.v[2*j+1] = (K + 1 < H_) ? tobf(lds_attn[(K + 1) * 16 + n]) : tobf(0.f);
      b.u[2*j] = 0; b.u[2*j+1] = 0;
    }
    // B (K=h, N=d): only lanes 0-15, dwords 0..5 carry h=0..11
    if (!half) {
      const unsigned* tp = (const unsigned*)(Tst + ((size_t)s * D_ + d) * 16);
      #pragma unroll
      for (int j = 0; j < 6; ++j) {
        unsigned w2 = tp[j];
        b.u[2*j]   = (unsigned short)(w2 & 0xFFFFu);
        b.u[2*j+1] = (unsigned short)(w2 >> 16);
      }
    }
    if (s + 1 < S_)   // prefetch next B row (global_prefetch_b8)
      __builtin_prefetch(Tst + ((size_t)(s + 1) * D_ + d) * 16, 0, 1);

    // analytic per-(k,s) mean of rwl row (lane computes for k = k0 + n)
    float mk = 0.f;
    #pragma unroll
    for (int h = 0; h < H_; ++h) mk += lds_attn[h * 16 + n] * meanT[h * S_ + s];
    if (k0 + n == s) mk += meanH[s];

    v8f c = {};
    c = __builtin_amdgcn_wmma_f32_16x16x32_bf16(false, a.v, false, b.v, (short)0, c, false, false);

    #pragma unroll
    for (int r = 0; r < 8; ++r) {
      int k = k0 + r + 8 * half;
      float vv = c[r];
      if (k == s) vv += hidden[(size_t)s * D_ + d];        // residual diagonal
      float mv = __shfl(mk, r + 8 * half, 32);             // broadcast mean for this k
      float tv = (vv - mv) * lnw;
      acc[r] += tv;                                        // attn_output accumulation
      out_tvstd[((size_t)k * S_ + s) * D_ + d] = tv * invs[r];
    }
  }
  #pragma unroll
  for (int r = 0; r < 8; ++r)
    ws_attnout[(size_t)(k0 + r + 8 * half) * D_ + d] = acc[r];
}

// resultant[k,d] = (attn_output[k,d] + bias_term[d]) / std[k] + ln_bias[d]
__global__ void k_resultant(const float* __restrict__ attn_out, const float* __restrict__ bias_term,
                            const float* __restrict__ inv_std, const float* __restrict__ ln_b,
                            float* __restrict__ res) {
  int idx = blockIdx.x * blockDim.x + threadIdx.x;
  if (idx >= S_ * D_) return;
  int k = idx / D_, dd = idx % D_;
  res[idx] = (attn_out[idx] + bias_term[dd]) * inv_std[k] + ln_b[dd];
}

// importance + tv_norm: wave per (k,s) row of tv_std
__global__ void k_final(const float* __restrict__ tvstd, const float* __restrict__ res,
                        float* __restrict__ imp, float* __restrict__ nrm) {
  int lane = threadIdx.x & 31;
  int row  = blockIdx.x * (blockDim.x >> 5) + (threadIdx.x >> 5);
  if (row >= S_ * S_) return;
  int k = row / S_;
  const float* tp = tvstd + (size_t)row * D_;
  const float* rp = res + (size_t)k * D_;
  float n2 = 0.f, l1 = 0.f;
  for (int i = lane; i < D_; i += 32) {
    float t = tp[i];
    n2 += t * t;
    l1 += fabsf(t - rp[i] + 1e-6f);
  }
  #pragma unroll
  for (int m = 16; m; m >>= 1) { n2 += __shfl_xor(n2, m, 32); l1 += __shfl_xor(l1, m, 32); }
  if (lane == 0) { nrm[row] = sqrtf(n2); imp[row] = -l1; }
}

// ------------------------------------------------------------------------
extern "C" void kernel_launch(void* const* d_in, const int* in_sizes, int n_in,
                              void* d_out, int out_size, void* d_ws, size_t ws_size,
                              hipStream_t stream) {
  const float* val   = (const float*)d_in[0];   // value_layer     [H,S,V]
  const float* attn  = (const float*)d_in[1];   // attention_probs [H,S,S]
  const float* hid   = (const float*)d_in[2];   // hidden_states   [S,D]
  const float* preln = (const float*)d_in[3];   // pre_ln_states   [S,D]
  const float* w     = (const float*)d_in[4];   // dense_w         [D,H,V]
  const float* bias  = (const float*)d_in[5];   // dense_bias      [D]
  const float* lnw   = (const float*)d_in[6];   // ln_weight       [D]
  const float* lnb   = (const float*)d_in[7];   // ln_bias         [D]

  // workspace layout (~10.2 MB total)
  char* ws = (char*)d_ws;
  unsigned short* Tst = (unsigned short*)ws;           // [S][D][16] bf16 : 9,437,184 B
  float* wbar   = (float*)(ws + 9437184);              // 3,072 B
  float* meanT  = (float*)(ws + 9440256);              // 18,432 B
  float* meanH  = (float*)(ws + 9458688);              // 1,536 B
  float* invstd = (float*)(ws + 9460224);              // 1,536 B
  float* bterm  = (float*)(ws + 9461760);              // 3,072 B
  float* aout   = (float*)(ws + 9464832);              // 1,179,648 B

  // output layout: importance[S,S], tv_norm[S,S], tv_std[S,S,D], resultant[S,D]
  float* out   = (float*)d_out;
  float* o_imp = out;
  float* o_nrm = out + (size_t)S_ * S_;
  float* o_tvs = out + (size_t)2 * S_ * S_;
  float* o_res = out + (size_t)2 * S_ * S_ + (size_t)S_ * S_ * D_;

  k_wbar       <<<3, 256, 0, stream>>>(w, wbar);
  k_meanT      <<<18, 256, 0, stream>>>(val, wbar, meanT);
  k_rowmean    <<<48, 256, 0, stream>>>(hid, meanH);
  k_invstd     <<<48, 256, 0, stream>>>(preln, invstd);
  k_bias       <<<1, 256, 0, stream>>>(bias, lnw, bterm);
  k_transformed<<<dim3(H_, 6), 256, 0, stream>>>(val, w, Tst);
  k_main       <<<dim3(S_ / 16, 6), 256, 0, stream>>>(attn, hid, Tst, meanT, meanH,
                                                      invstd, lnw, o_tvs, aout);
  k_resultant  <<<(S_ * D_ + 255) / 256, 256, 0, stream>>>(aout, bterm, invstd, lnb, o_res);
  k_final      <<<(S_ * S_) / 8, 256, 0, stream>>>(o_tvs, o_res, o_imp, o_nrm);
}
